// FastSelfAttention_3616362463292
// MI455X (gfx1250) — compile-verified, matrix operand order checked by reading, not run on hardware
//
#include <hip/hip_runtime.h>
#include <hip/hip_bf16.h>

// ---------------------------------------------------------------------------
// FastSelfAttention (Fastformer additive attention) for gfx1250 / MI455X
// B=4, S=4096, H=1024, NH=16, HS=64.
// GEMMs via v_wmma_f32_16x16x32_bf16; A tiles DMA'd by the Tensor Data Mover
// (tensor_load_to_lds + s_wait_tensorcnt) when the builtin is available.
// ---------------------------------------------------------------------------

#define BATCH   4
#define SEQ     4096
#define HDIM    1024
#define NHEADS  16
#define HSIZE   64
#define MROWS   (BATCH * SEQ)          // 16384
#define ELEMS   ((size_t)MROWS * HDIM) // 16,777,216

typedef __attribute__((ext_vector_type(16))) __bf16 v16bf;
typedef __attribute__((ext_vector_type(8)))  __bf16 v8bf;
typedef __attribute__((ext_vector_type(8)))  float  v8f;
typedef __attribute__((ext_vector_type(4)))  unsigned int v4ui;
typedef __attribute__((ext_vector_type(8)))  int v8si;
typedef __attribute__((ext_vector_type(4)))  int v4si;

union BFrag { v16bf v; v8bf h[2]; };

#if defined(__AMDGCN__) && __has_builtin(__builtin_amdgcn_tensor_load_to_lds) && \
    __has_builtin(__builtin_amdgcn_s_wait_tensorcnt)
#define USE_TDM 1
#else
#define USE_TDM 0
#endif

__device__ __forceinline__ unsigned short f2bfu(float f) {
    unsigned u = __builtin_bit_cast(unsigned, f);
    return (unsigned short)((u + 0x7FFFu + ((u >> 16) & 1u)) >> 16); // RNE
}
__device__ __forceinline__ unsigned pack2bf(float a, float b) {
    return (unsigned)f2bfu(a) | ((unsigned)f2bfu(b) << 16);
}

// ---------------------------------------------------------------------------
// Weight conversion: f32 -> bf16, 4 elements / thread
// ---------------------------------------------------------------------------
__global__ __launch_bounds__(256) void wcvt_kernel(const float* __restrict__ in,
                                                   __bf16* __restrict__ out) {
    size_t i = ((size_t)blockIdx.x * 256 + threadIdx.x) * 4;
    float4 v = *(const float4*)(in + i);
    uint2 p;
    p.x = pack2bf(v.x, v.y);
    p.y = pack2bf(v.z, v.w);
    *(uint2*)(out + i) = p;
}

// ---------------------------------------------------------------------------
// LayerNorm1: one wave32 per row of 1024. Writes f32 + bf16 copies.
// ---------------------------------------------------------------------------
__global__ __launch_bounds__(256) void ln1_kernel(const float* __restrict__ X,
                                                  const float* __restrict__ g,
                                                  const float* __restrict__ bt,
                                                  float* __restrict__ Y,
                                                  __bf16* __restrict__ Ybf) {
    const int lane = threadIdx.x & 31;
    const int wav  = threadIdx.x >> 5;
    const size_t row = (size_t)blockIdx.x * 8 + wav;
    const float* x = X + row * HDIM;

    float4 v[8];
    float s = 0.f, s2 = 0.f;
#pragma unroll
    for (int i = 0; i < 8; ++i) {
        v[i] = *(const float4*)(x + (size_t)(i * 32 + lane) * 4);
        s  += v[i].x + v[i].y + v[i].z + v[i].w;
        s2 += v[i].x * v[i].x + v[i].y * v[i].y + v[i].z * v[i].z + v[i].w * v[i].w;
    }
#pragma unroll
    for (int off = 16; off; off >>= 1) {
        s  += __shfl_xor(s,  off, 32);
        s2 += __shfl_xor(s2, off, 32);
    }
    const float mean = s * (1.f / HDIM);
    const float var  = s2 * (1.f / HDIM) - mean * mean;
    const float rstd = rsqrtf(var + 1e-5f);

#pragma unroll
    for (int i = 0; i < 8; ++i) {
        const int c = (i * 32 + lane) * 4;
        float4 gg = *(const float4*)(g  + c);
        float4 bb = *(const float4*)(bt + c);
        float4 o;
        o.x = (v[i].x - mean) * rstd * gg.x + bb.x;
        o.y = (v[i].y - mean) * rstd * gg.y + bb.y;
        o.z = (v[i].z - mean) * rstd * gg.z + bb.z;
        o.w = (v[i].w - mean) * rstd * gg.w + bb.w;
        *(float4*)(Y + row * HDIM + c) = o;
        uint2 p;
        p.x = pack2bf(o.x, o.y);
        p.y = pack2bf(o.z, o.w);
        *(uint2*)(Ybf + row * HDIM + c) = p;
    }
}

// ---------------------------------------------------------------------------
// GEMM (NT): C[M,N] = A[M,K](bf16) * W[N,K](bf16)^T, f32 accumulate.
// Block tile 256x64, 8 waves; each wave owns 2 M-tiles x 4 N-tiles
// (8 wmma / K-step, B fragments reused across M).  Double-buffered LDS.
// A tile staged by the Tensor Data Mover when available.
// ---------------------------------------------------------------------------
#define TMBLK 256
#define TN    64
#define TK    32

#if USE_TDM
// Build a D# for a 2D tile load: tile TMBLK x TK bf16 out of an M x K
// row-major bf16 tensor, destination LDS offset lds_off. (ISA 8.3/8.4)
__device__ __forceinline__ void tdm_load_a_tile(const __bf16* gsrc,
                                                unsigned lds_off,
                                                int M, int K) {
    unsigned long long ga = (unsigned long long)(const void*)gsrc;
    v4ui g0 = {0u, 0u, 0u, 0u};
    g0[0] = 1u;                                   // count=1 valid descriptor
    g0[1] = lds_off;                              // lds_addr
    g0[2] = (unsigned)ga;                         // global_addr[31:0]
    g0[3] = (unsigned)((ga >> 32) & 0x1FFFFFFu)   // global_addr[56:32]
            | (2u << 30);                         // type = 2 ("image")
    v8si g1 = {0, 0, 0, 0, 0, 0, 0, 0};
    g1[0] = 1 << 16;                              // data_size = 2 bytes
    g1[1] = (int)(((unsigned)K & 0xFFFFu) << 16);                 // tensor_dim0 lo
    g1[2] = (int)((((unsigned)K >> 16) & 0xFFFFu) |
                  (((unsigned)M & 0xFFFFu) << 16));               // dim0 hi | dim1 lo
    g1[3] = (int)((((unsigned)M >> 16) & 0xFFFFu) |
                  ((unsigned)TK << 16));                          // dim1 hi | tile_dim0
    g1[4] = TMBLK;                                // tile_dim1 (rows), tile_dim2=0
    g1[5] = K;                                    // tensor_dim0_stride lo32
    v4si z4 = {0, 0, 0, 0};
#if __clang_major__ >= 23
    v8si z8 = {0, 0, 0, 0, 0, 0, 0, 0};
    __builtin_amdgcn_tensor_load_to_lds(g0, g1, z4, z4, z8, 0);
#else
    __builtin_amdgcn_tensor_load_to_lds(g0, g1, z4, z4, 0);
#endif
}
#endif

__global__ __launch_bounds__(256) void gemm_bf16_nt(const __bf16* __restrict__ A,
                                                    const __bf16* __restrict__ W,
                                                    float* __restrict__ C,
                                                    int M, int N, int K) {
    __shared__ __align__(16) __bf16 As[2][TMBLK][TK];   // 2 x 16 KB
    __shared__ __align__(16) __bf16 Bs[2][TN][TK];      // 2 x  4 KB

    const int lane = threadIdx.x & 31;
    const int wav  = threadIdx.x >> 5;
    const int m0   = blockIdx.x * TMBLK;
    const int n0   = blockIdx.y * TN;

    const int lhalf = lane >> 4;   // which K-octet group this lane holds
    const int mn    = lane & 15;   // row (A) / col (B) index inside 16
    const int kb    = lhalf * 8;   // K base within first 16

    const int br = threadIdx.x >> 2;        // B-tile row (= output col)
    const int bc = (threadIdx.x & 3) * 8;   // 8 bf16 per thread

    v8f acc[2][4] = {};

#if USE_TDM
    typedef __attribute__((address_space(3))) const void* lds_cptr_t;
    const unsigned lds_a0 = (unsigned)(unsigned long long)(lds_cptr_t)(&As[0][0][0]);
    const unsigned lds_a1 = (unsigned)(unsigned long long)(lds_cptr_t)(&As[1][0][0]);
#endif

    const int nsteps = K / TK;

    // ---- prologue: stage tile 0 into buffer 0
    {
        uint4 b0 = *(const uint4*)(W + (size_t)(n0 + br) * K + bc);
        *(uint4*)(&Bs[0][br][bc]) = b0;
#if USE_TDM
        if (wav == 0) {
            tdm_load_a_tile(A + (size_t)m0 * K, lds_a0, M, K);
            __builtin_amdgcn_s_wait_tensorcnt(0);
        }
#else
#pragma unroll
        for (int j = 0; j < 4; ++j) {
            uint4 a0 = *(const uint4*)(A + (size_t)(m0 + threadIdx.x) * K + j * 8);
            *(uint4*)(&As[0][threadIdx.x][j * 8]) = a0;
        }
#endif
    }
    __syncthreads();

    for (int i = 0; i < nsteps; ++i) {
        const int cur = i & 1;
        const int nxt = cur ^ 1;
        const int kn  = (i + 1) * TK;
        const bool more = (i + 1) < nsteps;

        // ---- kick off fetch of tile i+1 (overlaps with compute below)
        uint4 bnext = {0u, 0u, 0u, 0u};
#if !USE_TDM
        uint4 anext[4];
#endif
        if (more) {
            bnext = *(const uint4*)(W + (size_t)(n0 + br) * K + kn + bc);
#if USE_TDM
            if (wav == 0)
                tdm_load_a_tile(A + (size_t)m0 * K + kn, nxt ? lds_a1 : lds_a0, M, K);
#else
#pragma unroll
            for (int j = 0; j < 4; ++j)
                anext[j] = *(const uint4*)(A + (size_t)(m0 + threadIdx.x) * K + kn + j * 8);
#endif
        }

        // ---- compute on buffer `cur`: 2 M-tiles x 4 N-tiles
        BFrag a0, a1;
        const int r0 = wav * 32 + mn;
        a0.h[0] = *(const v8bf*)(&As[cur][r0][kb]);
        a0.h[1] = *(const v8bf*)(&As[cur][r0][kb + 16]);
        a1.h[0] = *(const v8bf*)(&As[cur][r0 + 16][kb]);
        a1.h[1] = *(const v8bf*)(&As[cur][r0 + 16][kb + 16]);
#pragma unroll
        for (int nt = 0; nt < 4; ++nt) {
            BFrag b;
            const int bcol = nt * 16 + mn;
            b.h[0] = *(const v8bf*)(&Bs[cur][bcol][kb]);
            b.h[1] = *(const v8bf*)(&Bs[cur][bcol][kb + 16]);
            acc[0][nt] = __builtin_amdgcn_wmma_f32_16x16x32_bf16(
                false, a0.v, false, b.v, (short)0, acc[0][nt], false, false);
            acc[1][nt] = __builtin_amdgcn_wmma_f32_16x16x32_bf16(
                false, a1.v, false, b.v, (short)0, acc[1][nt], false, false);
        }

        // ---- publish tile i+1
        if (more) {
            *(uint4*)(&Bs[nxt][br][bc]) = bnext;
#if USE_TDM
            if (wav == 0) __builtin_amdgcn_s_wait_tensorcnt(0);
#else
#pragma unroll
            for (int j = 0; j < 4; ++j)
                *(uint4*)(&As[nxt][threadIdx.x][j * 8]) = anext[j];
#endif
        }
        __syncthreads();
    }

    // C/D layout: VGPR v -> M = v + 8*(lane/16), N = lane%16
#pragma unroll
    for (int mt = 0; mt < 2; ++mt) {
#pragma unroll
        for (int nt = 0; nt < 4; ++nt) {
#pragma unroll
            for (int vv = 0; vv < 8; ++vv) {
                const int row = m0 + wav * 32 + mt * 16 + vv + lhalf * 8;
                const int col = n0 + nt * 16 + mn;
                C[(size_t)row * N + col] = acc[mt][nt][vv];
            }
        }
    }
}

// ---------------------------------------------------------------------------
// Additive-attention pooling (causal running mean weighted by exp logits).
// One wave32 per (batch, head); lane owns 2 of the 64 channels.
// Writes pooled result as bf16 (feeds the Wp GEMM).
// ---------------------------------------------------------------------------
__global__ __launch_bounds__(32) void pool_bf16_kernel(const float* __restrict__ X,
                                                       const float* __restrict__ vecw,
                                                       const int* __restrict__ mask,
                                                       __bf16* __restrict__ Obf) {
    const int b  = blockIdx.x >> 4;
    const int hd = blockIdx.x & 15;
    const int lane = threadIdx.x;
    const int c = hd * HSIZE + lane * 2;
    const float2 va = *(const float2*)(vecw + c);

    float2 num = {0.f, 0.f};
    float den = 0.f;
    for (int s = 0; s < SEQ; ++s) {
        const size_t idx = ((size_t)b * SEQ + s) * HDIM + c;
        const float2 x = *(const float2*)(X + idx);
        float p = x.x * va.x + x.y * va.y;
#pragma unroll
        for (int off = 16; off; off >>= 1) p += __shfl_xor(p, off, 32);
        const float m = (1.f - (float)mask[b * SEQ + s]) * -10000.f;
        const float a = __expf(p * 0.125f + m);   // 1/sqrt(64)
        num.x += a * x.x;
        num.y += a * x.y;
        den   += a;
        const float inv = 1.f / den;
        *(unsigned*)(Obf + idx) = pack2bf(num.x * inv, num.y * inv);
    }
}

// Same scan, fused final epilogue: out = residual + pooled * values
__global__ __launch_bounds__(32) void pool_final_kernel(const float* __restrict__ X,
                                                        const float* __restrict__ vecw,
                                                        const int* __restrict__ mask,
                                                        const float* __restrict__ Res,
                                                        const float* __restrict__ Val,
                                                        float* __restrict__ Out) {
    const int b  = blockIdx.x >> 4;
    const int hd = blockIdx.x & 15;
    const int lane = threadIdx.x;
    const int c = hd * HSIZE + lane * 2;
    const float2 va = *(const float2*)(vecw + c);

    float2 num = {0.f, 0.f};
    float den = 0.f;
    for (int s = 0; s < SEQ; ++s) {
        const size_t idx = ((size_t)b * SEQ + s) * HDIM + c;
        const float2 x = *(const float2*)(X + idx);
        float p = x.x * va.x + x.y * va.y;
#pragma unroll
        for (int off = 16; off; off >>= 1) p += __shfl_xor(p, off, 32);
        const float m = (1.f - (float)mask[b * SEQ + s]) * -10000.f;
        const float a = __expf(p * 0.125f + m);
        num.x += a * x.x;
        num.y += a * x.y;
        den   += a;
        const float inv = 1.f / den;
        const float2 r = *(const float2*)(Res + idx);
        const float2 v = *(const float2*)(Val + idx);
        float2 o;
        o.x = r.x + num.x * inv * v.x;
        o.y = r.y + num.y * inv * v.y;
        *(float2*)(Out + idx) = o;
    }
}

// ---------------------------------------------------------------------------
// residual = hn + pq*keys (overwrites hn); mk = LN2(residual)
// ---------------------------------------------------------------------------
__global__ __launch_bounds__(256) void mix_ln2_kernel(float* __restrict__ HN,
                                                      const float* __restrict__ PQ,
                                                      const float* __restrict__ KY,
                                                      const float* __restrict__ g,
                                                      const float* __restrict__ bt,
                                                      float* __restrict__ MK) {
    const int lane = threadIdx.x & 31;
    const int wav  = threadIdx.x >> 5;
    const size_t row = (size_t)blockIdx.x * 8 + wav;
    const size_t base = row * HDIM;

    float4 r[8];
    float s = 0.f, s2 = 0.f;
#pragma unroll
    for (int i = 0; i < 8; ++i) {
        const int c = (i * 32 + lane) * 4;
        float4 hv = *(const float4*)(HN + base + c);
        float4 qv = *(const float4*)(PQ + base + c);
        float4 kv = *(const float4*)(KY + base + c);
        r[i].x = hv.x + qv.x * kv.x;
        r[i].y = hv.y + qv.y * kv.y;
        r[i].z = hv.z + qv.z * kv.z;
        r[i].w = hv.w + qv.w * kv.w;
        *(float4*)(HN + base + c) = r[i];     // residual in place
        s  += r[i].x + r[i].y + r[i].z + r[i].w;
        s2 += r[i].x * r[i].x + r[i].y * r[i].y + r[i].z * r[i].z + r[i].w * r[i].w;
    }
#pragma unroll
    for (int off = 16; off; off >>= 1) {
        s  += __shfl_xor(s,  off, 32);
        s2 += __shfl_xor(s2, off, 32);
    }
    const float mean = s * (1.f / HDIM);
    const float var  = s2 * (1.f / HDIM) - mean * mean;
    const float rstd = rsqrtf(var + 1e-5f);

#pragma unroll
    for (int i = 0; i < 8; ++i) {
        const int c = (i * 32 + lane) * 4;
        float4 gg = *(const float4*)(g  + c);
        float4 bb = *(const float4*)(bt + c);
        float4 o;
        o.x = (r[i].x - mean) * rstd * gg.x + bb.x;
        o.y = (r[i].y - mean) * rstd * gg.y + bb.y;
        o.z = (r[i].z - mean) * rstd * gg.z + bb.z;
        o.w = (r[i].w - mean) * rstd * gg.w + bb.w;
        *(float4*)(MK + base + c) = o;
    }
}

// ---------------------------------------------------------------------------
extern "C" void kernel_launch(void* const* d_in, const int* in_sizes, int n_in,
                              void* d_out, int out_size, void* d_ws, size_t ws_size,
                              hipStream_t stream) {
    (void)in_sizes; (void)n_in; (void)out_size; (void)ws_size;

    const float* hidden = (const float*)d_in[0];
    const int*   amask  = (const int*)d_in[1];
    const float* Wq     = (const float*)d_in[2];
    const float* Wk     = (const float*)d_in[3];
    const float* Wv     = (const float*)d_in[4];
    const float* Wp     = (const float*)d_in[5];
    const float* qattn  = (const float*)d_in[6];
    const float* kattn  = (const float*)d_in[7];
    const float* g1     = (const float*)d_in[8];
    const float* b1     = (const float*)d_in[9];
    const float* g2     = (const float*)d_in[10];
    const float* b2     = (const float*)d_in[11];
    float* out = (float*)d_out;

    char* ws = (char*)d_ws;
    const size_t F = ELEMS * sizeof(float);         // 64 MB
    float*  hn  = (float*)(ws);                     // LN1 out, then residual
    float*  Qb  = (float*)(ws + 1 * F);             // query, later mk
    float*  Kb  = (float*)(ws + 2 * F);             // keys
    float*  Vb  = (float*)(ws + 3 * F);             // values
    float*  PQ  = (float*)(ws + 4 * F);             // pooled_query @ Wp^T
    __bf16* Abf = (__bf16*)(ws + 5 * F);            // bf16 A operand (32 MB)
    __bf16* Wqb = (__bf16*)(ws + 5 * F + ELEMS * sizeof(__bf16));
    __bf16* Wkb = Wqb + (size_t)HDIM * HDIM;
    __bf16* Wvb = Wkb + (size_t)HDIM * HDIM;
    __bf16* Wpb = Wvb + (size_t)HDIM * HDIM;

    const int wgrid = (HDIM * HDIM) / (256 * 4);    // 1024 blocks
    wcvt_kernel<<<wgrid, 256, 0, stream>>>(Wq, Wqb);
    wcvt_kernel<<<wgrid, 256, 0, stream>>>(Wk, Wkb);
    wcvt_kernel<<<wgrid, 256, 0, stream>>>(Wv, Wvb);
    wcvt_kernel<<<wgrid, 256, 0, stream>>>(Wp, Wpb);

    ln1_kernel<<<MROWS / 8, 256, 0, stream>>>(hidden, g1, b1, hn, Abf);

    dim3 ggrid(MROWS / TMBLK, HDIM / TN);           // 64 x 16
    gemm_bf16_nt<<<ggrid, 256, 0, stream>>>(Abf, Wqb, Qb, MROWS, HDIM, HDIM);
    gemm_bf16_nt<<<ggrid, 256, 0, stream>>>(Abf, Wkb, Kb, MROWS, HDIM, HDIM);
    gemm_bf16_nt<<<ggrid, 256, 0, stream>>>(Abf, Wvb, Vb, MROWS, HDIM, HDIM);

    // pooled_query (bf16 into Abf; hn-bf16 no longer needed)
    pool_bf16_kernel<<<BATCH * NHEADS, 32, 0, stream>>>(Qb, qattn, amask, Abf);

    // pq = pooled_query @ Wp^T
    gemm_bf16_nt<<<ggrid, 256, 0, stream>>>(Abf, Wpb, PQ, MROWS, HDIM, HDIM);

    // residual (in hn) + LN2 -> mk (reuse Qb)
    mix_ln2_kernel<<<MROWS / 8, 256, 0, stream>>>(hn, PQ, Kb, g2, b2, Qb);

    // out = residual + pooled_keys * values
    pool_final_kernel<<<BATCH * NHEADS, 32, 0, stream>>>(Qb, kattn, amask, hn, Vb, out);
}
